// RoIHead_20882130993682
// MI455X (gfx1250) — matrix operand order repaired and checked by reading, not compiled, not run on hardware
//
#include <hip/hip_runtime.h>

typedef __attribute__((ext_vector_type(16))) _Float16 v16h;
typedef __attribute__((ext_vector_type(8)))  _Float16 v8h;
typedef __attribute__((ext_vector_type(8)))  float    v8f;
typedef __attribute__((ext_vector_type(4)))  unsigned int v4u;
typedef __attribute__((ext_vector_type(8)))  int v8i;
typedef __attribute__((ext_vector_type(4)))  int v4i;

#define RR    64
#define GRID3 216
#define MPTS  (RR*GRID3)   // 13824
#define NPTS  4096
#define CIN   32
#define CG    35           // 3 + 32
#define NS    16
#define PRE   27648        // 216 * 128
#define F1    256

union U16h { v16h v; v8h v8[2]; _Float16 h[16]; };
union U8f  { v8f v; float f[8]; };

__device__ inline v8f wmma_f16(v16h a, v16h b, v8f c) {
  return __builtin_amdgcn_wmma_f32_16x16x32_f16(false, a, false, b, (short)0, c, false, false);
}
__device__ inline v8f zero8() { v8f z = {0.f,0.f,0.f,0.f,0.f,0.f,0.f,0.f}; return z; }

// ---- Tensor Data Mover: 1-D DMA of `ndwords` dwords from global -> LDS ----
// D# group0: count=1, lds_addr, global_addr[56:0], type=2.  group1: data_size=4B,
// tensor_dim0=tile_dim0=ndwords, tensor_dim1=1, tile_dim1=1.  Groups 2/3 zero.
__device__ inline void tdm_load_1d(unsigned lds_addr, const void* gptr, unsigned ndwords) {
  unsigned long long ga = (unsigned long long)gptr;
  v4u g0;
  g0[0] = 1u;                                             // count=1
  g0[1] = lds_addr;                                       // bits[63:32] lds addr
  g0[2] = (unsigned)(ga & 0xffffffffu);                   // global_addr[31:0]
  g0[3] = (unsigned)((ga >> 32) & 0x01ffffffu) | (2u<<30); // global_addr[56:32] | type=2
  v8i g1;
  g1[0] = (int)(2u << 16);                                // wg_mask=0, data_size=2 (4B)
  g1[1] = (int)((ndwords & 0xffffu) << 16);               // tensor_dim0[15:0]
  g1[2] = (int)(((ndwords >> 16) & 0xffffu) | (1u << 16)); // tensor_dim0[31:16] | tensor_dim1[15:0]=1
  g1[3] = (int)((ndwords & 0xffffu) << 16);               // tensor_dim1[31:16]=0 | tile_dim0
  g1[4] = 1;                                              // tile_dim1=1, tile_dim2=0
  g1[5] = (int)ndwords;                                   // tensor_dim0_stride[31:0]
  g1[6] = 0; g1[7] = 0;
  v4i z4 = {0,0,0,0};
#if __clang_major__ >= 23
  v8i z8 = {0,0,0,0,0,0,0,0};
  __builtin_amdgcn_tensor_load_to_lds(g0, g1, z4, z4, z8, 0);
#else
  __builtin_amdgcn_tensor_load_to_lds(g0, g1, z4, z4, 0);
#endif
}
__device__ inline unsigned lds_addr_of(const void* p) {
  return (unsigned)(unsigned long long)(uintptr_t)p;   // low 32 bits of flat addr = LDS byte offset
}

// ---------------- prep: f32 -> f16 fragment-packed weights ----------------

// wsmall packed layout (halfs): idx = ((((branch*2+mat)*2 + kc)*4 + nt)*32 + lane)*16 + i
// mat0 = w1 (35x64 zero-padded to 64x64), mat1 = w2 (64x64); KxN row-major sources.
__global__ void k_prep_small(const float* __restrict__ w1a, const float* __restrict__ w2a,
                             const float* __restrict__ w1b, const float* __restrict__ w2b,
                             _Float16* __restrict__ wsmall) {
  int t = blockIdx.x * blockDim.x + threadIdx.x;
  if (t >= 4 * 4096) return;
  int i = t & 15, lane = (t >> 4) & 31, nt = (t >> 9) & 3, kc = (t >> 11) & 1;
  int mat = (t >> 12) & 1, branch = (t >> 13) & 1;
  int n = nt*16 + (lane & 15);
  int k = kc*32 + ((lane >= 16) ? 8 : 0) + (i & 7) + ((i >= 8) ? 16 : 0);
  const float* w1 = branch ? w1b : w1a;
  const float* w2 = branch ? w2b : w2a;
  float v = (mat == 0) ? ((k < CG) ? w1[k*64 + n] : 0.f) : w2[k*64 + n];
  wsmall[t] = (_Float16)v;
}

// ws1 packed into B-fragment order: packed[((kc*16+nt)*32+lane)*16+i]
__global__ void k_pack_ws1(const float* __restrict__ ws1, _Float16* __restrict__ packed) {
  const int total = (PRE/32) * 16 * 32 * 16;   // 7,077,888
  for (int e = blockIdx.x * blockDim.x + threadIdx.x; e < total; e += gridDim.x * blockDim.x) {
    int i    = e & 15;
    int lane = (e >> 4) & 31;
    int nt   = (e >> 9) & 15;
    int kc   = e >> 13;
    int n = nt*16 + (lane & 15);
    int k = kc*32 + ((lane >= 16) ? 8 : 0) + (i & 7) + ((i >= 8) ? 16 : 0);
    packed[e] = (_Float16)ws1[k*F1 + n];
  }
}

// 7 x (256x256) fc weights -> f16 fragment-packed, contiguous: ws2,wc1,wc2,wi1,wi2,wr1,wr2
__global__ void k_prep_fc(const float* ws2, const float* wc1, const float* wc2,
                          const float* wi1, const float* wi2, const float* wr1, const float* wr2,
                          _Float16* __restrict__ dst) {
  int t = blockIdx.x * blockDim.x + threadIdx.x;
  if (t >= 7 * 65536) return;
  int mat = t >> 16, e = t & 65535;
  const float* src;
  switch (mat) {
    case 0: src = ws2; break; case 1: src = wc1; break; case 2: src = wc2; break;
    case 3: src = wi1; break; case 4: src = wi2; break; case 5: src = wr1; break;
    default: src = wr2; break;
  }
  int i = e & 15, lane = (e >> 4) & 31, nt = (e >> 9) & 15, kc = (e >> 13) & 7;
  int n = nt*16 + (lane & 15);
  int k = kc*32 + ((lane >= 16) ? 8 : 0) + (i & 7) + ((i >= 8) ? 16 : 0);
  dst[t] = (_Float16)src[k*256 + n];
}

// ---------------- stage 1: grid points ----------------
__global__ void k_gridpoints(const float* __restrict__ rois, float* __restrict__ nxyz) {
  int m = blockIdx.x * blockDim.x + threadIdx.x;
  if (m >= MPTS) return;
  int r = m / GRID3, p = m % GRID3;
  int i = p / 36, j = (p / 6) % 6, k = p % 6;
  const float* ro = rois + r*7;
  float sx = ro[3], sy = ro[4], sz = ro[5];
  float lx = ((float)i + 0.5f) * (1.0f/6.0f) * sx - 0.5f * sx;
  float ly = ((float)j + 0.5f) * (1.0f/6.0f) * sy - 0.5f * sy;
  float lz = ((float)k + 0.5f) * (1.0f/6.0f) * sz - 0.5f * sz;
  float cr = cosf(ro[6]), sr = sinf(ro[6]);
  nxyz[m*3 + 0] = lx*cr - ly*sr + ro[0];
  nxyz[m*3 + 1] = lx*sr + ly*cr + ro[1];
  nxyz[m*3 + 2] = lz + ro[2];
}

// ---------------- stage 2: ball query (TDM stages whole cloud into LDS) ----------------
__global__ __launch_bounds__(256) void k_ballquery(const float* __restrict__ nxyz,
                                                   const float* __restrict__ xyz,
                                                   int* __restrict__ idxA, int* __restrict__ idxB,
                                                   int* __restrict__ cntA, int* __restrict__ cntB) {
  __shared__ float sP[NPTS*3];   // 48 KB of the 320 KB WGP LDS, interleaved xyz
  if (threadIdx.x == 0) {
    tdm_load_1d(lds_addr_of(sP), xyz, NPTS*3);   // one DMA, TENSORcnt tracked
    __builtin_amdgcn_s_wait_tensorcnt(0);
  }
  __syncthreads();
  int m = blockIdx.x * 256 + threadIdx.x;
  if (m >= MPTS) return;
  float cx = nxyz[m*3], cy = nxyz[m*3+1], cz = nxyz[m*3+2];
  const float r0 = 0.8f*0.8f, r1 = 1.6f*1.6f;
  int ca = 0, cb = 0, fa = 0, fb = 0;
  for (int j = 0; j < NPTS; ++j) {
    float dx = sP[j*3+0]-cx, dy = sP[j*3+1]-cy, dz = sP[j*3+2]-cz;
    float d2 = dx*dx + dy*dy + dz*dz;
    if (d2 < r1) {
      if (cb < NS) { if (cb == 0) fb = j; idxB[m*NS + cb] = j; cb++; }
      if (d2 < r0 && ca < NS) { if (ca == 0) fa = j; idxA[m*NS + ca] = j; ca++; }
      if (ca == NS) break;   // r0-hits are a subset of r1-hits -> cb==NS too
    }
  }
  for (int t = ca; t < NS; ++t) idxA[m*NS + t] = fa;
  for (int t = cb; t < NS; ++t) idxB[m*NS + t] = fb;
  cntA[m] = ca; cntB[m] = cb;
}

// ---------------- stage 3: gather + 2-layer MLP + maxpool (WMMA, 1 wave / (point,branch)) ----------------
__global__ __launch_bounds__(256) void k_group(const float* __restrict__ nxyz,
                                               const float* __restrict__ xyz,
                                               const float* __restrict__ feats,
                                               const int* __restrict__ idxA, const int* __restrict__ idxB,
                                               const int* __restrict__ cntA, const int* __restrict__ cntB,
                                               const _Float16* __restrict__ wsmall,
                                               _Float16* __restrict__ pooled) {
  __shared__ _Float16 sw[4*4096];          // fragment-packed weights, 32 KB (via TDM)
  __shared__ _Float16 apack[8][2*32*16];   // per-wave layer-1 output in A-fragment order, 16 KB
  if (threadIdx.x == 0) {
    tdm_load_1d(lds_addr_of(sw), wsmall, (4*4096*2)/4);
    __builtin_amdgcn_s_wait_tensorcnt(0);
  }
  __syncthreads();

  int lane = threadIdx.x & 31;
  int wv   = threadIdx.x >> 5;
  int gw   = blockIdx.x * 8 + wv;
  int m = gw >> 1, branch = gw & 1;

  const int* idxp = branch ? idxB : idxA;
  int cnt = branch ? cntB[m] : cntA[m];
  const _Float16* W1 = sw + (branch*2 + 0)*4096;   // [kc][nt][lane][16]
  const _Float16* W2 = sw + (branch*2 + 1)*4096;

  int s   = lane & 15;                       // sample row this lane feeds
  int kb8 = (lane >= 16) ? 8 : 0;            // f16 A-layout K sub-offset
  int pid = idxp[m*NS + s];
  float cx = nxyz[m*3], cy = nxyz[m*3+1], cz = nxyz[m*3+2];
  float rel0 = xyz[pid*3+0] - cx, rel1 = xyz[pid*3+1] - cy, rel2 = xyz[pid*3+2] - cz;
  const float* fr = feats + pid*CIN;

  auto g_at = [&](int k) -> float {
    if (cnt == 0 || k >= CG) return 0.f;
    if (k == 0) return rel0;
    if (k == 1) return rel1;
    if (k == 2) return rel2;
    return fr[k-3];
  };

  U16h A0, A1;   // K chunk 0 (0..31) and chunk 1 (32..63, zero-pad past 35)
  #pragma unroll
  for (int i = 0; i < 8; ++i) {
    A0.h[i]   = (_Float16)g_at(kb8 + i);
    A0.h[8+i] = (_Float16)g_at(16 + kb8 + i);
    A1.h[i]   = (_Float16)g_at(32 + kb8 + i);
    A1.h[8+i] = (_Float16)g_at(48 + kb8 + i);
  }

  auto bfrag = [&](const _Float16* W, int nt, int kc) -> v16h {
    U16h u;
    const _Float16* p = W + ((kc*4 + nt)*32 + lane)*16;  // contiguous 32B per lane
    u.v8[0] = *(const v8h*)(p);
    u.v8[1] = *(const v8h*)(p + 8);
    return u.v;
  };

  // layer 1: h1 = relu(g @ W1); scatter results directly into A-fragment order in LDS
  int colb = lane & 15;
  int rowb = (lane >= 16) ? 8 : 0;
  #pragma unroll
  for (int t = 0; t < 4; ++t) {
    v8f c = zero8();
    c = wmma_f16(A0.v, bfrag(W1, t, 0), c);
    c = wmma_f16(A1.v, bfrag(W1, t, 1), c);
    U8f u; u.v = c;
    #pragma unroll
    for (int r = 0; r < 8; ++r) {
      int row  = rowb + r;
      int colK = t*16 + colb;
      int within = colK & 31, kc2 = colK >> 5;
      int lane_t = (row & 15) + (((within >> 3) & 1) ? 16 : 0);
      int slot   = (within & 7) + ((within >= 16) ? 8 : 0);
      apack[wv][(kc2*32 + lane_t)*16 + slot] = (_Float16)fmaxf(u.f[r], 0.f);
    }
  }
  // per-wave LDS region; same-wave DS ops are in-order -> no barrier needed

  // layer 2: h2 = relu(h1 @ W2); maxpool over 16 sample rows (relu commutes with max)
  int roi = m / GRID3, p = m % GRID3;
  #pragma unroll
  for (int t = 0; t < 4; ++t) {
    v8f c = zero8();
    #pragma unroll
    for (int kc = 0; kc < 2; ++kc) {
      U16h a;
      const _Float16* ap = &apack[wv][(kc*32 + lane)*16];
      a.v8[0] = *(const v8h*)(ap);
      a.v8[1] = *(const v8h*)(ap + 8);
      c = wmma_f16(a.v, bfrag(W2, t, kc), c);
    }
    U8f u; u.v = c;
    float tm = u.f[0];
    #pragma unroll
    for (int r = 1; r < 8; ++r) tm = fmaxf(tm, u.f[r]);
    tm = fmaxf(tm, __shfl_xor(tm, 16, 32));   // combine M=r and M=r+8 half-waves
    tm = fmaxf(tm, 0.f);
    if (lane < 16) {
      int ch = branch*64 + t*16 + lane;
      pooled[(size_t)roi*PRE + ch*GRID3 + p] = (_Float16)tm;   // pre-transposed layout
    }
  }
}

// ---------------- stage 4/5: C[64xN=256] = (relu)(A[64xK] @ Bpacked[KxN]) ----------------
__global__ __launch_bounds__(256) void k_gemm(const _Float16* __restrict__ A,
                                              const _Float16* __restrict__ Bp,
                                              _Float16* __restrict__ C, int K, int relu) {
  int lane  = threadIdx.x & 31;
  int wave  = blockIdx.x * 8 + (threadIdx.x >> 5);
  int mtile = wave >> 4, ntile = wave & 15;
  int kb8   = (lane >= 16) ? 8 : 0;
  int kchunks = K >> 5;
  const _Float16* Ar = A + (size_t)(mtile*16 + (lane & 15)) * K;
  v8f c = zero8();
  for (int kc = 0; kc < kchunks; ++kc) {
    U16h a, b;
    a.v8[0] = *(const v8h*)(Ar + kc*32 + kb8);
    a.v8[1] = *(const v8h*)(Ar + kc*32 + 16 + kb8);
    b.v = *(const v16h*)(Bp + ((size_t)(kc*16 + ntile)*32 + lane)*16);
    __builtin_prefetch(Bp + ((size_t)((kc+1)*16 + ntile)*32 + lane)*16, 0, 1); // next chunk
    c = wmma_f16(a.v, b.v, c);
  }
  U8f u; u.v = c;
  int col = ntile*16 + (lane & 15);
  int rb  = (lane >= 16) ? 8 : 0;
  #pragma unroll
  for (int r = 0; r < 8; ++r) {
    float v = u.f[r];
    if (relu) v = fmaxf(v, 0.f);
    C[(mtile*16 + rb + r)*F1 + col] = (_Float16)v;
  }
}

// ---------------- stage 6: head projections (256 -> 1/1/7) + bias, f32 out ----------------
__global__ void k_heads(const _Float16* __restrict__ hc, const _Float16* __restrict__ hi,
                        const _Float16* __restrict__ hr,
                        const float* wc3, const float* bc3, const float* wi3, const float* bi3,
                        const float* wr3, const float* br3, float* __restrict__ out) {
  int t = blockIdx.x * blockDim.x + threadIdx.x;
  if (t >= RR*9) return;
  int r = t / 9, o = t % 9;
  const _Float16* h; const float* w; float bias; int stride, col;
  if (o == 0)      { h = hc; w = wc3; bias = bc3[0]; stride = 1; col = 0; }
  else if (o == 1) { h = hi; w = wi3; bias = bi3[0]; stride = 1; col = 0; }
  else             { h = hr; w = wr3; bias = br3[o-2]; stride = 7; col = o-2; }
  float acc = 0.f;
  for (int k = 0; k < 256; ++k) acc += (float)h[r*256 + k] * w[k*stride + col];
  out[t] = acc + bias;
}

extern "C" void kernel_launch(void* const* d_in, const int* in_sizes, int n_in,
                              void* d_out, int out_size, void* d_ws, size_t ws_size,
                              hipStream_t stream) {
  (void)in_sizes; (void)n_in; (void)out_size; (void)ws_size;
  const float* rois  = (const float*)d_in[0];
  const float* xyz   = (const float*)d_in[1];
  const float* feats = (const float*)d_in[2];
  const float* w1a = (const float*)d_in[3];
  const float* w2a = (const float*)d_in[4];
  const float* w1b = (const float*)d_in[5];
  const float* w2b = (const float*)d_in[6];
  const float* ws1 = (const float*)d_in[7];
  const float* ws2 = (const float*)d_in[8];
  const float* wc1 = (const float*)d_in[9];
  const float* wc2 = (const float*)d_in[10];
  const float* wc3 = (const float*)d_in[11];
  const float* bc3 = (const float*)d_in[12];
  const float* wi1 = (const float*)d_in[13];
  const float* wi2 = (const float*)d_in[14];
  const float* wi3 = (const float*)d_in[15];
  const float* bi3 = (const float*)d_in[16];
  const float* wr1 = (const float*)d_in[17];
  const float* wr2 = (const float*)d_in[18];
  const float* wr3 = (const float*)d_in[19];
  const float* br3 = (const float*)d_in[20];

  char* ws = (char*)d_ws;
  size_t off = 0;
  auto suballoc = [&](size_t bytes) -> char* {
    char* p = ws + off;
    off = (off + bytes + 255) & ~((size_t)255);
    return p;
  };
  _Float16* wsmall = (_Float16*)suballoc(4*4096*sizeof(_Float16));
  _Float16* ws1p   = (_Float16*)suballoc((size_t)PRE*F1*sizeof(_Float16));
  _Float16* wfc    = (_Float16*)suballoc((size_t)7*65536*sizeof(_Float16));
  float*    nxyz   = (float*)suballoc((size_t)MPTS*3*sizeof(float));
  int* idxA = (int*)suballoc((size_t)MPTS*NS*sizeof(int));
  int* idxB = (int*)suballoc((size_t)MPTS*NS*sizeof(int));
  int* cntA = (int*)suballoc((size_t)MPTS*sizeof(int));
  int* cntB = (int*)suballoc((size_t)MPTS*sizeof(int));
  _Float16* pooled = (_Float16*)suballoc((size_t)RR*PRE*sizeof(_Float16));
  _Float16* sh1 = (_Float16*)suballoc((size_t)RR*F1*sizeof(_Float16));
  _Float16* sh2 = (_Float16*)suballoc((size_t)RR*F1*sizeof(_Float16));
  _Float16* hc1 = (_Float16*)suballoc((size_t)RR*F1*sizeof(_Float16));
  _Float16* hc2 = (_Float16*)suballoc((size_t)RR*F1*sizeof(_Float16));
  _Float16* hi1 = (_Float16*)suballoc((size_t)RR*F1*sizeof(_Float16));
  _Float16* hi2 = (_Float16*)suballoc((size_t)RR*F1*sizeof(_Float16));
  _Float16* hr1 = (_Float16*)suballoc((size_t)RR*F1*sizeof(_Float16));
  _Float16* hr2 = (_Float16*)suballoc((size_t)RR*F1*sizeof(_Float16));

  k_prep_small<<<64, 256, 0, stream>>>(w1a, w2a, w1b, w2b, wsmall);
  k_pack_ws1<<<4096, 256, 0, stream>>>(ws1, ws1p);
  k_prep_fc<<<(7*65536 + 255)/256, 256, 0, stream>>>(ws2, wc1, wc2, wi1, wi2, wr1, wr2, wfc);
  k_gridpoints<<<(MPTS + 255)/256, 256, 0, stream>>>(rois, nxyz);
  k_ballquery<<<MPTS/256, 256, 0, stream>>>(nxyz, xyz, idxA, idxB, cntA, cntB);
  k_group<<<(MPTS*2)/8, 256, 0, stream>>>(nxyz, xyz, feats, idxA, idxB, cntA, cntB, wsmall, pooled);
  k_gemm<<<8, 256, 0, stream>>>(pooled, ws1p, sh1, PRE, 1);
  k_gemm<<<8, 256, 0, stream>>>(sh1, wfc + 0*65536, sh2, F1, 1);
  k_gemm<<<8, 256, 0, stream>>>(sh2, wfc + 1*65536, hc1, F1, 1);
  k_gemm<<<8, 256, 0, stream>>>(hc1, wfc + 2*65536, hc2, F1, 1);
  k_gemm<<<8, 256, 0, stream>>>(sh2, wfc + 3*65536, hi1, F1, 1);
  k_gemm<<<8, 256, 0, stream>>>(hi1, wfc + 4*65536, hi2, F1, 1);
  k_gemm<<<8, 256, 0, stream>>>(sh2, wfc + 5*65536, hr1, F1, 1);
  k_gemm<<<8, 256, 0, stream>>>(hr1, wfc + 6*65536, hr2, F1, 1);
  k_heads<<<(RR*9 + 255)/256, 256, 0, stream>>>(hc2, hi2, hr2, wc3, bc3, wi3, bi3, wr3, br3,
                                                (float*)d_out);
}
